// DynamicCrossVariableFilter_80599356276761
// MI455X (gfx1250) — matrix operand order repaired and checked by reading, not compiled, not run on hardware
//
#include <hip/hip_runtime.h>
#include <hip/hip_bf16.h>

#define BDIM 64
#define CDIM 128
#define DDIM 2048

typedef __attribute__((ext_vector_type(16))) __bf16 v16bf;
typedef __attribute__((ext_vector_type(8)))  float  v8f;

__device__ __forceinline__ unsigned short f2bf(float f) {
    unsigned int u = __float_as_uint(f);
    unsigned int r = u + 0x7FFFu + ((u >> 16) & 1u);   // round-to-nearest-even
    return (unsigned short)(r >> 16);
}
__device__ __forceinline__ float bf2f(unsigned short h) {
    return __uint_as_float(((unsigned int)h) << 16);
}

// 32-byte fragment load via two 16B-aligned uint4 loads (LDS rows are 16B aligned,
// not 32B, so avoid over-declaring alignment through a v16bf*).
__device__ __forceinline__ v16bf load16(const unsigned short* p) {
    union { uint4 q[2]; v16bf v; } u;
    u.q[0] = ((const uint4*)p)[0];
    u.q[1] = ((const uint4*)p)[1];
    return u.v;
}

// ---------------------------------------------------------------------------
// Kernel 1: exact 0.9-quantile of |x| along D per (b,c) row (bitonic sort in LDS)
// ---------------------------------------------------------------------------
__global__ __launch_bounds__(256) void quantile_kernel(const float* __restrict__ x,
                                                       float* __restrict__ thr) {
    __shared__ float s[DDIM];
    int row = blockIdx.x;                         // b*C + c
    const float* xr = x + (size_t)row * DDIM * 2;
    for (int i = threadIdx.x; i < DDIM; i += 256) {
        float re = xr[2 * i], im = xr[2 * i + 1];
        s[i] = sqrtf(re * re + im * im);
    }
    __syncthreads();
    for (int k = 2; k <= DDIM; k <<= 1) {
        for (int j = k >> 1; j > 0; j >>= 1) {
            for (int i = threadIdx.x; i < DDIM; i += 256) {
                int ixj = i ^ j;
                if (ixj > i) {
                    bool up = ((i & k) == 0);
                    float a = s[i], b2 = s[ixj];
                    if ((a > b2) == up) { s[i] = b2; s[ixj] = a; }
                }
            }
            __syncthreads();
        }
    }
    if (threadIdx.x == 0) {
        const double pos = 0.9 * (double)(DDIM - 1);  // 1842.3
        int i0 = (int)pos;
        float frac = (float)(pos - (double)i0);
        thr[row] = s[i0] + frac * (s[i0 + 1] - s[i0]);
    }
}

// ---------------------------------------------------------------------------
// Kernel 2: normed_weights = softmax(relu(weights)) per component, over axis=1
// ---------------------------------------------------------------------------
__global__ __launch_bounds__(CDIM) void softmax_kernel(const float* __restrict__ w,
                                                       float* __restrict__ wr,
                                                       float* __restrict__ wi) {
    __shared__ float red[CDIM * 2];
    int i = blockIdx.x, j = threadIdx.x;
    float re = fmaxf(w[(i * CDIM + j) * 2], 0.0f);
    float im = fmaxf(w[(i * CDIM + j) * 2 + 1], 0.0f);
    red[j] = re; red[CDIM + j] = im;
    __syncthreads();
    for (int s2 = CDIM / 2; s2 > 0; s2 >>= 1) {
        if (j < s2) {
            red[j]        = fmaxf(red[j],        red[j + s2]);
            red[CDIM + j] = fmaxf(red[CDIM + j], red[CDIM + j + s2]);
        }
        __syncthreads();
    }
    float mr = red[0], mi = red[CDIM];
    __syncthreads();
    float er = expf(re - mr), ei = expf(im - mi);
    red[j] = er; red[CDIM + j] = ei;
    __syncthreads();
    for (int s2 = CDIM / 2; s2 > 0; s2 >>= 1) {
        if (j < s2) {
            red[j]        += red[j + s2];
            red[CDIM + j] += red[CDIM + j + s2];
        }
        __syncthreads();
    }
    wr[i * CDIM + j] = er / red[0];
    wi[i * CDIM + j] = ei / red[CDIM];
}

// ---------------------------------------------------------------------------
// Kernel 3: pack A fragments (16-bit A 16x32 lane layout, ISA 7.12.2) as
// bf16 hi/lo planes: 0=Wr_hi 1=Wr_lo 2=Wi_hi 3=Wi_lo 4=-Wi_hi 5=-Wi_lo
// ---------------------------------------------------------------------------
#define PSTRIDE (32 * 32 * 16)   // frags(8 mt * 4 ks) * 32 lanes * 16 elems

__global__ __launch_bounds__(32) void pack_kernel(const float* __restrict__ wr,
                                                  const float* __restrict__ wi,
                                                  unsigned short* __restrict__ afr) {
    int mt   = blockIdx.x >> 2;      // 0..7
    int ks   = blockIdx.x & 3;       // 0..3
    int lane = threadIdx.x;          // 0..31
    int hf   = lane >> 4;
    int m    = mt * 16 + (lane & 15);
    int fragbase = (mt * 4 + ks) * (32 * 16);
    for (int e = 0; e < 16; ++e) {
        // element e of the 16-bit A fragment -> local K 16*hf + e (contiguous, matches B)
        int kl = 16 * hf + e;
        int k  = ks * 32 + kl;
        float vr = wr[m * CDIM + k];
        float vi = wi[m * CDIM + k];
        unsigned short rh = f2bf(vr);
        unsigned short rl = f2bf(vr - bf2f(rh));
        unsigned short ih = f2bf(vi);
        unsigned short il = f2bf(vi - bf2f(ih));
        int off = fragbase + lane * 16 + e;
        afr[0 * PSTRIDE + off] = rh;
        afr[1 * PSTRIDE + off] = rl;
        afr[2 * PSTRIDE + off] = ih;
        afr[3 * PSTRIDE + off] = il;
        afr[4 * PSTRIDE + off] = ih ^ 0x8000u;   // exact bf16 negate
        afr[5 * PSTRIDE + off] = il ^ 0x8000u;
    }
}

// ---------------------------------------------------------------------------
// Kernel 4: fused mask + complex matmul (bf16x2 split via WMMA) + epilogue
// ---------------------------------------------------------------------------
#define WMMA(A, Bm, Cacc) \
    Cacc = __builtin_amdgcn_wmma_f32_16x16x32_bf16(false, A, false, Bm, (short)0, Cacc, false, false)

__global__ __launch_bounds__(256) void main_kernel(const float* __restrict__ x,
                                                   const float* __restrict__ amp,
                                                   const float* __restrict__ mixp,
                                                   const float* __restrict__ thr,
                                                   const unsigned short* __restrict__ afr,
                                                   float* __restrict__ out) {
    constexpr int NB    = 32;             // d-columns per block
    constexpr int PITCH = 136;            // ushorts per n-row (272B, 16B aligned)
    constexpr int PLANE = NB * PITCH;     // 4352 ushorts (8704B, 16B aligned)
    __shared__ unsigned int smem_u32[(4 * PLANE) / 2];     // 34816 bytes
    unsigned short* sm = (unsigned short*)smem_u32;

    int blk = blockIdx.x;
    int b   = blk >> 6;                   // / (DDIM/NB)
    int d0  = (blk & 63) * NB;
    int tid = threadIdx.x;

    // Stage conj(masked signal) transposed [n=d][k=c], bf16 hi/lo planes.
    for (int it = 0; it < 16; ++it) {
        int idx = tid + 256 * it;         // 0..4095
        int c   = idx >> 5;
        int dl  = idx & 31;
        float2 v = *(const float2*)(x + 2 * ((size_t)(b * CDIM + c) * DDIM + d0 + dl));
        float t = thr[b * CDIM + c];
        bool keep = sqrtf(v.x * v.x + v.y * v.y) > t;
        float sr = keep ? v.x : 0.0f;
        float si = keep ? -v.y : 0.0f;    // conjugate
        unsigned short sh = f2bf(sr);
        unsigned short sl = f2bf(sr - bf2f(sh));
        unsigned short th = f2bf(si);
        unsigned short tl = f2bf(si - bf2f(th));
        int o = dl * PITCH + c;
        sm[0 * PLANE + o] = sh;
        sm[1 * PLANE + o] = sl;
        sm[2 * PLANE + o] = th;
        sm[3 * PLANE + o] = tl;
    }
    __syncthreads();

    int wave = tid >> 5;                  // M-tile 0..7
    int lane = tid & 31;
    int hf   = lane >> 4;
    int n    = lane & 15;

    v8f accre[2] = {}, accim[2] = {};

    for (int ks = 0; ks < 4; ++ks) {
        int aoff = ((wave * 4 + ks) * 32 + lane) * 16;
        v16bf wrh = load16(afr + 0 * PSTRIDE + aoff);
        v16bf wrl = load16(afr + 1 * PSTRIDE + aoff);
        v16bf wih = load16(afr + 2 * PSTRIDE + aoff);
        v16bf wil = load16(afr + 3 * PSTRIDE + aoff);
        v16bf nih = load16(afr + 4 * PSTRIDE + aoff);
        v16bf nil = load16(afr + 5 * PSTRIDE + aoff);
#pragma unroll
        for (int nt = 0; nt < 2; ++nt) {
            // B fragment: lane = column n, elems = K 16*hf..16*hf+15 (contiguous)
            int bo = (nt * 16 + n) * PITCH + ks * 32 + hf * 16;
            v16bf srh = load16(sm + 0 * PLANE + bo);
            v16bf srl = load16(sm + 1 * PLANE + bo);
            v16bf sih = load16(sm + 2 * PLANE + bo);
            v16bf sil = load16(sm + 3 * PLANE + bo);
            // re += Wr*Sr - Wi*Si   (3-term hi/lo products each)
            WMMA(wrh, srh, accre[nt]);
            WMMA(wrh, srl, accre[nt]);
            WMMA(wrl, srh, accre[nt]);
            WMMA(nih, sih, accre[nt]);
            WMMA(nih, sil, accre[nt]);
            WMMA(nil, sih, accre[nt]);
            // im += Wr*Si + Wi*Sr
            WMMA(wrh, sih, accim[nt]);
            WMMA(wrh, sil, accim[nt]);
            WMMA(wrl, sih, accim[nt]);
            WMMA(wih, srh, accim[nt]);
            WMMA(wih, srl, accim[nt]);
            WMMA(wil, srh, accim[nt]);
        }
    }

    // Epilogue: out = (1-mix)*x + mix*((x*amp)*wsum)
#pragma unroll
    for (int nt = 0; nt < 2; ++nt) {
#pragma unroll
        for (int r = 0; r < 8; ++r) {
            int cro = wave * 16 + hf * 8 + r;     // C/D layout: VGPR r, half-wave
            int d   = d0 + nt * 16 + n;
            size_t gi = (size_t)(b * CDIM + cro) * DDIM + d;
            float2 xv = *(const float2*)(x + 2 * gi);
            float a  = amp[cro * DDIM + d];
            float2 m = *(const float2*)(mixp + 2 * cro);
            float wsr = accre[nt][r], wsi = accim[nt][r];
            float fr = a * (xv.x * wsr - xv.y * wsi);
            float fi = a * (xv.x * wsi + xv.y * wsr);
            float omr = 1.0f - m.x, omi = -m.y;
            float2 o;
            o.x = omr * xv.x - omi * xv.y + m.x * fr - m.y * fi;
            o.y = omr * xv.y + omi * xv.x + m.x * fi + m.y * fr;
            *(float2*)(out + 2 * gi) = o;
        }
    }
}

extern "C" void kernel_launch(void* const* d_in, const int* in_sizes, int n_in,
                              void* d_out, int out_size, void* d_ws, size_t ws_size,
                              hipStream_t stream) {
    (void)in_sizes; (void)n_in; (void)out_size; (void)ws_size;
    const float* x   = (const float*)d_in[0];   // complex64 interleaved [B,C,D]
    const float* amp = (const float*)d_in[1];   // float32 [C,D]
    const float* w   = (const float*)d_in[2];   // complex64 interleaved [C,C]
    const float* mix = (const float*)d_in[3];   // complex64 interleaved [C,1]
    float* out = (float*)d_out;

    char* ws = (char*)d_ws;
    float* thr = (float*)(ws);                              // 8192 f  = 32 KB
    float* wr  = (float*)(ws + 32768);                      // 16384 f = 64 KB
    float* wi  = (float*)(ws + 32768 + 65536);              // 64 KB
    unsigned short* afr = (unsigned short*)(ws + 32768 + 2 * 65536); // 192 KB

    quantile_kernel<<<BDIM * CDIM, 256, 0, stream>>>(x, thr);
    softmax_kernel<<<CDIM, CDIM, 0, stream>>>(w, wr, wi);
    pack_kernel<<<32, 32, 0, stream>>>(wr, wi, afr);
    main_kernel<<<BDIM * (DDIM / 32), 256, 0, stream>>>(x, amp, mix, thr, afr, out);
}
